// GM_RNN_79577154060905
// MI455X (gfx1250) — matrix operand order repaired and checked by reading, not compiled, not run on hardware
//
#include <hip/hip_runtime.h>
#include <stdint.h>

// Problem dims (match reference)
#define TT 512
#define BB 64
#define XX 64
#define HH 1024
#define YY 64
#define DD 2

typedef _Float16 v16h __attribute__((ext_vector_type(16)));
typedef _Float16 v8h  __attribute__((ext_vector_type(8)));
typedef float    v8f  __attribute__((ext_vector_type(8)));
typedef long long i64;

// ---------------------------------------------------------------------------
// WMMA helpers (CDNA5 gfx1250, wave32). D = A(16x32 f16) * B(32x16 f16) + C(f32)
// ---------------------------------------------------------------------------
__device__ __forceinline__ v8f wmma32(v16h a, v16h b, v8f c) {
  return __builtin_amdgcn_wmma_f32_16x16x32_f16(false, a, false, b, (short)0, c,
                                                false, false);
}

// B fragment: 32x16 tile of W^T, i.e. rows n0..n0+15 of row-major W[N][K].
// Contiguous k order (same as A): lane l<16 holds column n=n0+l, K=k0..k0+15; lane>=16 holds the
// same column at K=k0+16..k0+31.  One 32B (2x b128) load per lane.
__device__ __forceinline__ v16h load_bfrag(const _Float16* __restrict__ W,
                                           int n0, int k0, int lane, int ldk) {
  int n = n0 + (lane & 15);
  int k = k0 + ((lane >> 4) << 4);
  return *(const v16h*)(W + (i64)n * ldk + k);
}

// A fragment: 16x32 tile of row-major f16 activations A[M][K].
// Contiguous k order (same as B): lane l<16 = row m0+l, K = k0..k0+15;
// lane>=16 = same rows, K = k0+16..k0+31. Two b128 loads.
__device__ __forceinline__ v16h load_afrag(const _Float16* __restrict__ A,
                                           i64 m0, int k0, int lane, int ldk) {
  i64 m = m0 + (lane & 15);
  int k = k0 + ((lane >> 4) << 4);
  const _Float16* p = A + m * (i64)ldk + k;
  v8h lo = *(const v8h*)p;
  v8h hi = *(const v8h*)(p + 8);
  return __builtin_shufflevector(lo, hi, 0, 1, 2, 3, 4, 5, 6, 7, 8, 9, 10, 11,
                                 12, 13, 14, 15);
}

// A fragment straight from f32 memory (encoder input x), converting inline.
__device__ __forceinline__ v16h load_afrag_f32(const float* __restrict__ A,
                                               i64 m0, int k0, int lane,
                                               int ldk) {
  i64 m = m0 + (lane & 15);
  int k = k0 + ((lane >> 4) << 4);
  const float* p = A + m * (i64)ldk + k;
  v16h a;
#pragma unroll
  for (int i = 0; i < 8; ++i) {
    a[i]     = (_Float16)p[i];
    a[i + 8] = (_Float16)p[i + 8];
  }
  return a;
}

// Branch-free fast transcendentals (single v_exp_f32, no exec divergence --
// libm tanhf lowers to a branchy ocml expansion which is poison on the
// serial critical path).
__device__ __forceinline__ float sigmoid_fast(float x) {
  return 1.0f / (1.0f + __expf(-x));
}
__device__ __forceinline__ float tanh_fast(float x) {
  float t = __expf(-2.0f * __builtin_fabsf(x));   // in (0,1]
  float y = (1.0f - t) / (1.0f + t);              // |tanh(x)|
  return __builtin_copysignf(y, x);
}

// ---------------------------------------------------------------------------
// Utility kernels
// ---------------------------------------------------------------------------
__global__ void cvt_f32_f16_kernel(const float* __restrict__ s,
                                   _Float16* __restrict__ d, i64 n) {
  i64 i = (i64)blockIdx.x * blockDim.x + threadIdx.x;
  i64 stride = (i64)gridDim.x * blockDim.x;
  for (; i < n; i += stride) d[i] = (_Float16)s[i];
}

__global__ void zero_u32_kernel(uint32_t* __restrict__ p, i64 n) {
  i64 i = (i64)blockIdx.x * blockDim.x + threadIdx.x;
  i64 stride = (i64)gridDim.x * blockDim.x;
  for (; i < n; i += stride) p[i] = 0u;
}

// ---------------------------------------------------------------------------
// Encoder (parallel over all T*B rows): out = f16(tanh(x @ enc_w^T + enc_b))
// M = T*B = 32768, K = X = 64, N = H = 1024. 8 waves/block, 1 tile/wave.
// ---------------------------------------------------------------------------
__global__ __launch_bounds__(256) void encoder_kernel(
    const float* __restrict__ x,        // [T*B, X] f32
    const _Float16* __restrict__ Wenc,  // [H, X]  f16
    const float* __restrict__ benc,     // [H]
    _Float16* __restrict__ out)         // [T*B, H] f16
{
  const int wave = threadIdx.x >> 5, lane = threadIdx.x & 31;
  const i64 tile = (i64)blockIdx.x * 8 + wave;  // (T*B/16) * (H/16) tiles
  const i64 rt = tile >> 6;                     // row tile (/ (H/16))
  const int ct = (int)(tile & 63);              // col tile
  const i64 m0 = rt * 16;
  const int n0 = ct * 16;

  v8f acc = {};
#pragma unroll
  for (int k0 = 0; k0 < XX; k0 += 32) {
    v16h a = load_afrag_f32(x, m0, k0, lane, XX);
    v16h b = load_bfrag(Wenc, n0, k0, lane, XX);
    acc = wmma32(a, b, acc);
  }
  // C/D layout: lane l<16 -> rows m0+0..7, col n0+l; l>=16 -> rows m0+8..15.
  const int col = n0 + (lane & 15);
  const i64 rbase = m0 + ((lane >> 4) << 3);
#pragma unroll
  for (int e = 0; e < 8; ++e) {
    float v = tanh_fast(acc[e] + benc[col]);
    out[(rbase + e) * (i64)HH + col] = (_Float16)v;
  }
}

// ---------------------------------------------------------------------------
// Fused GRU step for one layer (the serial critical path).
// Grid: 64 blocks (one per 16-wide column block of H). Block: 192 = 6 waves.
// Wave w: gate g = w>>1 (r,z,n), source s = w&1 (input/hidden path).
// Each wave: 4 row-tiles x K=1024 GEMM -> LDS -> fused GRU pointwise.
// ---------------------------------------------------------------------------
__global__ __launch_bounds__(192) void gru_step_kernel(
    const _Float16* __restrict__ Ain,      // [B,H] f16 input-path activations
    const _Float16* __restrict__ Hprev_h,  // [B,H] f16 previous hidden
    const float* __restrict__ Hprev_f,     // [B,H] f32 previous hidden
    const _Float16* __restrict__ Wih,      // [3H,H] f16 (layer-resolved)
    const _Float16* __restrict__ Whh,      // [3H,H] f16
    const float* __restrict__ bih,         // [3H]
    const float* __restrict__ bhh,         // [3H]
    float* __restrict__ Hout_f,            // [B,H] f32
    _Float16* __restrict__ Hout_h)         // [B,H] f16
{
  __shared__ float sAcc[6][BB][16];  // 24 KB: [ (gate,src) ][ row ][ col ]

  const int tid = threadIdx.x;
  const int wave = tid >> 5, lane = tid & 31;
  const int jb = blockIdx.x;  // H column block: cols jb*16 .. jb*16+15
  const int g = wave >> 1;    // 0=r 1=z 2=n
  const int s = wave & 1;     // 0: gi (input path), 1: gh (hidden path)

  const _Float16* A = s ? Hprev_h : Ain;
  const _Float16* W = s ? Whh : Wih;
  const int n0 = g * HH + jb * 16;

  // Per-lane base of this wave's B-fragment column (one W row).
  const _Float16* wrow =
      W + (i64)(n0 + (lane & 15)) * HH + ((lane >> 4) << 4);

  v8f acc[4];
#pragma unroll
  for (int r = 0; r < 4; ++r) { v8f z = {}; acc[r] = z; }

  for (int k0 = 0; k0 < HH; k0 += 32) {
    // global_prefetch_b8: give the next K-step's weight bytes a head start
    // down the L2->L0 path (speculative; tail prefetch lands in-buffer).
    __builtin_prefetch(wrow + k0 + 32, 0, 1);
    v16h b = *(const v16h*)(wrow + k0);
#pragma unroll
    for (int rb = 0; rb < 4; ++rb) {
      v16h a = load_afrag(A, rb * 16, k0, lane, HH);
      acc[rb] = wmma32(a, b, acc[rb]);
    }
  }

  const int col = lane & 15;
  const int rbase = (lane >> 4) << 3;
#pragma unroll
  for (int rb = 0; rb < 4; ++rb)
#pragma unroll
    for (int e = 0; e < 8; ++e)
      sAcc[wave][rb * 16 + rbase + e][col] = acc[rb][e];

  __syncthreads();

  // GRU pointwise over the 64x16 column block (PyTorch GRUCell semantics).
  for (int i = tid; i < BB * 16; i += 192) {
    int row = i >> 4, c = i & 15;
    int cg = jb * 16 + c;  // column in [0,H)
    float ir = sAcc[0][row][c] + bih[0 * HH + cg];
    float hr = sAcc[1][row][c] + bhh[0 * HH + cg];
    float iz = sAcc[2][row][c] + bih[1 * HH + cg];
    float hz = sAcc[3][row][c] + bhh[1 * HH + cg];
    float in_ = sAcc[4][row][c] + bih[2 * HH + cg];
    float hn = sAcc[5][row][c] + bhh[2 * HH + cg];
    float r = sigmoid_fast(ir + hr);
    float z = sigmoid_fast(iz + hz);
    float n = tanh_fast(in_ + r * hn);
    float hp = Hprev_f[(i64)row * HH + cg];
    float h = (1.0f - z) * n + z * hp;
    Hout_f[(i64)row * HH + cg] = h;
    Hout_h[(i64)row * HH + cg] = (_Float16)h;
  }
}

// ---------------------------------------------------------------------------
// Decoder + heads, fused (parallel over all T*B rows after the scan).
// Block handles 32 rows: phase 1 computes rnn_out = tanh(h @ dec_w^T + b)
// into 64KB of LDS (f16); phase 2 runs both heads (K=1024 from LDS).
// ---------------------------------------------------------------------------
__global__ __launch_bounds__(256) void decoder_heads_kernel(
    const _Float16* __restrict__ Hseq,  // [T*B, H] f16 (h1 per timestep)
    const _Float16* __restrict__ Wdec,  // [H, H] f16
    const float* __restrict__ bdec,     // [H]
    const _Float16* __restrict__ Wexp, const float* __restrict__ bexp,
    const _Float16* __restrict__ Wlog, const float* __restrict__ blog,
    float* __restrict__ out_mean,       // [T*B, Y]
    float* __restrict__ out_var)        // [T*B, Y]
{
  __shared__ _Float16 sDec[32][HH];  // 64 KB staged rnn_out block

  const int wave = threadIdx.x >> 5, lane = threadIdx.x & 31;
  const i64 m0 = (i64)blockIdx.x * 32;  // global row base

  // -------- phase 1: decoder GEMM, 8 waves x 8 col-tiles x 2 row-tiles ----
  for (int ct8 = 0; ct8 < 8; ++ct8) {
    const int n0 = (wave * 8 + ct8) * 16;
    v8f acc0 = {}, acc1 = {};
    for (int k0 = 0; k0 < HH; k0 += 32) {
      v16h b = load_bfrag(Wdec, n0, k0, lane, HH);
      v16h a0 = load_afrag(Hseq, m0, k0, lane, HH);
      v16h a1 = load_afrag(Hseq, m0 + 16, k0, lane, HH);
      acc0 = wmma32(a0, b, acc0);
      acc1 = wmma32(a1, b, acc1);
    }
    const int col = n0 + (lane & 15);
    const int rb = (lane >> 4) << 3;
#pragma unroll
    for (int e = 0; e < 8; ++e) {
      sDec[rb + e][col]      = (_Float16)tanh_fast(acc0[e] + bdec[col]);
      sDec[16 + rb + e][col] = (_Float16)tanh_fast(acc1[e] + bdec[col]);
    }
  }
  __syncthreads();

  // -------- phase 2: heads (mean / log-var), A-fragments from LDS ---------
  const int hd = wave >> 2;       // 0 = exp head (mean), 1 = log head (var)
  const int h_n0 = (wave & 3) * 16;
  const _Float16* Wh = hd ? Wlog : Wexp;
  const float* bh = hd ? blog : bexp;

  v8f acc0 = {}, acc1 = {};
  for (int k0 = 0; k0 < HH; k0 += 32) {
    v16h b = load_bfrag(Wh, h_n0, k0, lane, HH);
    v16h a0 = load_afrag(&sDec[0][0], 0, k0, lane, HH);
    v16h a1 = load_afrag(&sDec[0][0], 16, k0, lane, HH);
    acc0 = wmma32(a0, b, acc0);
    acc1 = wmma32(a1, b, acc1);
  }
  float* o = hd ? out_var : out_mean;
  const int col = h_n0 + (lane & 15);
  const int rb = (lane >> 4) << 3;
#pragma unroll
  for (int e = 0; e < 8; ++e) {
    float v0 = acc0[e] + bh[col];
    float v1 = acc1[e] + bh[col];
    if (hd) { v0 = __expf(v0); v1 = __expf(v1); }
    o[(m0 + rb + e) * (i64)YY + col] = v0;
    o[(m0 + 16 + rb + e) * (i64)YY + col] = v1;
  }
}

// ---------------------------------------------------------------------------
// Host launch
// ---------------------------------------------------------------------------
extern "C" void kernel_launch(void* const* d_in, const int* in_sizes, int n_in,
                              void* d_out, int out_size, void* d_ws,
                              size_t ws_size, hipStream_t stream) {
  (void)in_sizes; (void)n_in; (void)out_size;

  const float* x     = (const float*)d_in[0];
  const float* enc_w = (const float*)d_in[1];
  const float* enc_b = (const float*)d_in[2];
  const float* w_ih  = (const float*)d_in[3];
  const float* w_hh  = (const float*)d_in[4];
  const float* b_ih  = (const float*)d_in[5];
  const float* b_hh  = (const float*)d_in[6];
  const float* dec_w = (const float*)d_in[7];
  const float* dec_b = (const float*)d_in[8];
  const float* exp_w = (const float*)d_in[9];
  const float* exp_b = (const float*)d_in[10];
  const float* log_w = (const float*)d_in[11];
  const float* log_b = (const float*)d_in[12];

  // ---- workspace carve-out (all offsets 256B aligned) ----
  size_t off = 0;
  auto take = [&](size_t bytes) {
    size_t o = off;
    off = (off + bytes + 255) & ~(size_t)255;
    return o;
  };
  char* ws = (char*)d_ws;
  const i64 nWih = (i64)DD * 3 * HH * HH;   // 6,291,456
  const i64 nDec = (i64)HH * HH;
  const i64 nEnc = (i64)HH * XX;
  const i64 nHead = (i64)YY * HH;
  const i64 nBH = (i64)BB * HH;

  size_t o_wih  = take((size_t)nWih * 2);
  size_t o_whh  = take((size_t)nWih * 2);
  size_t o_enc  = take((size_t)nEnc * 2);
  size_t o_dec  = take((size_t)nDec * 2);
  size_t o_exp  = take((size_t)nHead * 2);
  size_t o_log  = take((size_t)nHead * 2);
  size_t o_rnn  = take((size_t)TT * nBH * 2);        // rnn_in f16 [T*B,H]
  size_t o_hseq = take((size_t)(TT + 1) * nBH * 2);  // h1 f16, slot0 = zeros
  size_t o_h0h  = take((size_t)2 * nBH * 2);         // h0 f16 double buffer
  size_t o_h0f  = take((size_t)2 * nBH * 4);         // h0 f32 double buffer
  size_t o_h1f  = take((size_t)2 * nBH * 4);         // h1 f32 double buffer
  if (off > ws_size) return;  // workspace too small: refuse to write OOB

  _Float16* wih_h  = (_Float16*)(ws + o_wih);
  _Float16* whh_h  = (_Float16*)(ws + o_whh);
  _Float16* encw_h = (_Float16*)(ws + o_enc);
  _Float16* decw_h = (_Float16*)(ws + o_dec);
  _Float16* expw_h = (_Float16*)(ws + o_exp);
  _Float16* logw_h = (_Float16*)(ws + o_log);
  _Float16* rnn_h  = (_Float16*)(ws + o_rnn);
  _Float16* hseq   = (_Float16*)(ws + o_hseq);
  _Float16* h0h[2] = {(_Float16*)(ws + o_h0h), (_Float16*)(ws + o_h0h) + nBH};
  float*    h0f[2] = {(float*)(ws + o_h0f), (float*)(ws + o_h0f) + nBH};
  float*    h1f[2] = {(float*)(ws + o_h1f), (float*)(ws + o_h1f) + nBH};

  // ---- one-time (per launch) weight downconversion to f16 ----
  auto cvt = [&](const float* s, _Float16* d, i64 n) {
    int blocks = (int)((n + 2047) / 2048);
    if (blocks > 2048) blocks = 2048;
    cvt_f32_f16_kernel<<<blocks, 256, 0, stream>>>(s, d, n);
  };
  cvt(w_ih, wih_h, nWih);
  cvt(w_hh, whh_h, nWih);
  cvt(enc_w, encw_h, nEnc);
  cvt(dec_w, decw_h, nDec);
  cvt(exp_w, expw_h, nHead);
  cvt(log_w, logw_h, nHead);

  // ---- zero initial hidden state (h0/h1, and hseq slot 0) ----
  zero_u32_kernel<<<64, 256, 0, stream>>>((uint32_t*)h0h[0], nBH / 2);
  zero_u32_kernel<<<64, 256, 0, stream>>>((uint32_t*)h0f[0], nBH);
  zero_u32_kernel<<<64, 256, 0, stream>>>((uint32_t*)h1f[0], nBH);
  zero_u32_kernel<<<64, 256, 0, stream>>>((uint32_t*)hseq, nBH / 2);

  // ---- hoisted encoder: parallel over all T*B rows ----
  // tiles = (T*B/16) * (H/16) = 131072, 8 tiles per 256-thread block
  encoder_kernel<<<(TT * BB / 16) * (HH / 16) / 8, 256, 0, stream>>>(
      x, encw_h, enc_b, rnn_h);

  // ---- serial recurrence: 2 fused kernels per timestep ----
  for (int t = 0; t < TT; ++t) {
    const int p = t & 1, q = (t + 1) & 1;
    // layer 0: input = encoder output
    gru_step_kernel<<<HH / 16, 192, 0, stream>>>(
        rnn_h + (i64)t * nBH, h0h[p], h0f[p], wih_h, whh_h, b_ih, b_hh,
        h0f[q], h0h[q]);
    // layer 1: input = layer-0 new hidden; writes h1 into hseq slot t+1
    gru_step_kernel<<<HH / 16, 192, 0, stream>>>(
        h0h[q], hseq + (i64)t * nBH, h1f[p], wih_h + (i64)3 * HH * HH,
        whh_h + (i64)3 * HH * HH, b_ih + 3 * HH, b_hh + 3 * HH, h1f[q],
        hseq + (i64)(t + 1) * nBH);
  }

  // ---- hoisted decoder + heads: parallel over all T*B rows ----
  float* out_mean = (float*)d_out;
  float* out_var = out_mean + (i64)TT * BB * YY;
  decoder_heads_kernel<<<TT * BB / 32, 256, 0, stream>>>(
      hseq + nBH, decw_h, dec_b, expw_h, exp_b, logw_h, log_b, out_mean,
      out_var);
}
